// MultiHeadAttention_3350074491358
// MI455X (gfx1250) — matrix operand order repaired and checked by reading, not compiled, not run on hardware
//
#include <hip/hip_runtime.h>

typedef __attribute__((ext_vector_type(16))) __bf16 v16bf;
typedef __attribute__((ext_vector_type(8)))  __bf16 v8bf;
typedef __attribute__((ext_vector_type(4)))  __bf16 v4bf;
typedef __attribute__((ext_vector_type(8)))  float  v8f;

#define D_MODEL 1024
#define SEQ     2048
#define NBATCH  2
#define NH      16
#define DHEAD   64
#define ROWS    (NBATCH * SEQ)   // 4096

// ---------------------------------------------------------------- fragments
__device__ __forceinline__ v16bf frag_a(const __bf16* p) {
  // A-matrix lane layout: 16 contiguous k values per lane (half*16 offset by caller)
  v8bf lo = *(const v8bf*)(p);
  v8bf hi = *(const v8bf*)(p + 8);
  return __builtin_shufflevector(lo, hi, 0,1,2,3,4,5,6,7,8,9,10,11,12,13,14,15);
}
__device__ __forceinline__ v16bf frag_b(const __bf16* p) {
  // B-matrix lane layout: 16 contiguous k values per lane (half*16 offset by caller)
  v8bf lo = *(const v8bf*)(p);
  v8bf hi = *(const v8bf*)(p + 8);
  return __builtin_shufflevector(lo, hi, 0,1,2,3,4,5,6,7,8,9,10,11,12,13,14,15);
}
__device__ __forceinline__ v8f wmma_bf16(v16bf a, v16bf b, v8f c) {
  return __builtin_amdgcn_wmma_f32_16x16x32_bf16(false, a, false, b, (short)0, c, false, false);
}
__device__ __forceinline__ float redmax16(float v) {
  v = fmaxf(v, __shfl_xor(v, 1, 32));
  v = fmaxf(v, __shfl_xor(v, 2, 32));
  v = fmaxf(v, __shfl_xor(v, 4, 32));
  v = fmaxf(v, __shfl_xor(v, 8, 32));
  return v;
}
__device__ __forceinline__ float redsum16(float v) {
  v += __shfl_xor(v, 1, 32);
  v += __shfl_xor(v, 2, 32);
  v += __shfl_xor(v, 4, 32);
  v += __shfl_xor(v, 8, 32);
  return v;
}

// ------------------------------------------------ async global->LDS (CDNA5)
// Generic pointers to __shared__ carry the DS-relative byte offset in the low
// 32 bits (flat->LDS uses addr[31:0]); that is the VDST operand of the async op.
__device__ __forceinline__ unsigned lds_off(const void* p) {
  return (unsigned)(unsigned long long)p;
}
__device__ __forceinline__ void async_copy_b128(unsigned lds, const void* gptr) {
  asm volatile("global_load_async_to_lds_b128 %0, %1, off"
               :: "v"(lds), "v"(gptr) : "memory");
}
__device__ __forceinline__ void wait_async0() {
  asm volatile("s_wait_asynccnt 0x0" ::: "memory");
}

// ---------------------------------------------------------------- converts
__global__ __launch_bounds__(256) void cvt_bf16(const float* __restrict__ s,
                                                __bf16* __restrict__ d, int n) {
  int i = (blockIdx.x * 256 + threadIdx.x) * 4;
  if (i >= n) return;
  float4 v = *(const float4*)(s + i);
  v4bf o = { (__bf16)v.x, (__bf16)v.y, (__bf16)v.z, (__bf16)v.w };
  *(v4bf*)(d + i) = o;
}

// W[k][n] fp32 -> WT[n][k] bf16  (1024x1024)
__global__ __launch_bounds__(256) void cvt_bf16_T(const float* __restrict__ W,
                                                  __bf16* __restrict__ WT) {
  int idx = blockIdx.x * 256 + threadIdx.x;
  int k = idx >> 10, n = idx & 1023;
  WT[(size_t)n * D_MODEL + k] = (__bf16)W[idx];
}

// ---------------------------------------------------------------- GEMM
// C[4096,1024] = A[4096,1024] * WT^T + bias ; 128x128 tile, 8 waves of 64x32.
// Double-buffered LDS staged by GLOBAL_LOAD_ASYNC_TO_LDS_B128 (ASYNCcnt).
// mode: 0=Q ([b][h][s][dh] bf16), 1=K (same), 2=V transposed ([b][h][dh][s] bf16),
//       3=final fp32 row-major to d_out.
__global__ __launch_bounds__(256) void gemm_bf16(const __bf16* __restrict__ A,
                                                 const __bf16* __restrict__ WT,
                                                 const float* __restrict__ bias,
                                                 void* __restrict__ outp, int mode) {
  __shared__ __align__(16) __bf16 As[2][128][40];
  __shared__ __align__(16) __bf16 Bs[2][128][40];
  const int tid   = threadIdx.x;
  const int lane  = tid & 31;
  const int wid   = tid >> 5;
  const int waveM = wid >> 2;      // 0..1
  const int waveN = wid & 3;       // 0..3
  const int l16   = lane & 15;
  const int half  = lane >> 4;
  const int mblock = blockIdx.x;   // 32
  const int nblock = blockIdx.y;   // 8

  const v8f vzero = {0.f,0.f,0.f,0.f,0.f,0.f,0.f,0.f};
  v8f acc[4][2];
#pragma unroll
  for (int mt = 0; mt < 4; ++mt)
#pragma unroll
    for (int nt = 0; nt < 2; ++nt) acc[mt][nt] = vzero;

  // each thread owns two 16B chunks of each 128x32 tile (512 chunks/tile)
  const int r0 = tid >> 2,          c0 = (tid & 3) * 8;
  const int r1 = (tid + 256) >> 2,  c1 = ((tid + 256) & 3) * 8;
  const __bf16* gA0 = A  + (size_t)(mblock * 128 + r0) * D_MODEL + c0;
  const __bf16* gA1 = A  + (size_t)(mblock * 128 + r1) * D_MODEL + c1;
  const __bf16* gB0 = WT + (size_t)(nblock * 128 + r0) * D_MODEL + c0;
  const __bf16* gB1 = WT + (size_t)(nblock * 128 + r1) * D_MODEL + c1;

  auto stage = [&](int buf, int kbase) {
    async_copy_b128(lds_off(&As[buf][r0][c0]), gA0 + kbase);
    async_copy_b128(lds_off(&As[buf][r1][c1]), gA1 + kbase);
    async_copy_b128(lds_off(&Bs[buf][r0][c0]), gB0 + kbase);
    async_copy_b128(lds_off(&Bs[buf][r1][c1]), gB1 + kbase);
  };

  const int NK = D_MODEL / 32;     // 32
  stage(0, 0);
  for (int kt = 0; kt < NK; ++kt) {
    const int cur = kt & 1;
    wait_async0();                 // this wave's copies into buf 'cur' complete
    __syncthreads();               // all waves done writing 'cur' / reading 'cur^1'
    if (kt + 1 < NK) stage(cur ^ 1, (kt + 1) * 32);   // overlap with compute

    v16bf bfr[2];
#pragma unroll
    for (int nt = 0; nt < 2; ++nt)
      bfr[nt] = frag_b(&Bs[cur][waveN * 32 + nt * 16 + l16][half * 16]);
#pragma unroll
    for (int mt = 0; mt < 4; ++mt) {
      v16bf af = frag_a(&As[cur][waveM * 64 + mt * 16 + l16][half * 16]);
#pragma unroll
      for (int nt = 0; nt < 2; ++nt)
        acc[mt][nt] = wmma_bf16(af, bfr[nt], acc[mt][nt]);
    }
  }

  // epilogue: C layout lane -> (M = j + 8*half, N = l16)
#pragma unroll
  for (int mt = 0; mt < 4; ++mt) {
#pragma unroll
    for (int nt = 0; nt < 2; ++nt) {
      const int col = nblock * 128 + waveN * 32 + nt * 16 + l16;
      const float bv = bias[col];
#pragma unroll
      for (int j = 0; j < 8; ++j) {
        const int r = mblock * 128 + waveM * 64 + mt * 16 + j + half * 8;
        const float val = acc[mt][nt][j] + bv;
        if (mode == 3) {
          ((float*)outp)[(size_t)r * D_MODEL + col] = val;
        } else {
          const int b = r >> 11, s = r & 2047;
          const int h = col >> 6, dh = col & 63;
          const int bh = b * NH + h;
          size_t idx;
          if (mode == 2) idx = ((size_t)bh * DHEAD + dh) * SEQ + s;       // V^T
          else           idx = ((size_t)bh * SEQ + s) * DHEAD + dh;       // Q/K
          ((__bf16*)outp)[idx] = (__bf16)val;
        }
      }
    }
  }
}

// ---------------------------------------------------------------- attention
// grid (16 qblocks, 32 bh), 8 waves x 16 q-rows. Flash-style online softmax.
__global__ __launch_bounds__(256) void attn(const __bf16* __restrict__ Qb,
                                            const __bf16* __restrict__ Kb,
                                            const __bf16* __restrict__ Vt,
                                            __bf16* __restrict__ Yb) {
  __shared__ __align__(16) __bf16 Ps[8][16][136];   // per-wave P tile (16 x 128)
  const int tid  = threadIdx.x;
  const int lane = tid & 31;
  const int wid  = tid >> 5;
  const int l16  = lane & 15;
  const int half = lane >> 4;
  const int qb   = blockIdx.x;     // 0..15
  const int bh   = blockIdx.y;     // 0..31

  const __bf16* Q = Qb + (size_t)bh * SEQ * DHEAD;
  const __bf16* K = Kb + (size_t)bh * SEQ * DHEAD;
  const __bf16* V = Vt + (size_t)bh * DHEAD * SEQ;
  const int qbase = qb * 128 + wid * 16;

  // Q fragments (A layout), pre-scaled by 1/sqrt(64)=0.125 (exact in bf16)
  v16bf qf[2];
#pragma unroll
  for (int s = 0; s < 2; ++s) {
    const __bf16* p = Q + (size_t)(qbase + l16) * DHEAD + s * 32 + half * 16;
    qf[s] = frag_a(p);
#pragma unroll
    for (int e = 0; e < 16; ++e) qf[s][e] = (__bf16)((float)qf[s][e] * 0.125f);
  }

  const v8f vzero = {0.f,0.f,0.f,0.f,0.f,0.f,0.f,0.f};
  v8f o[4];
#pragma unroll
  for (int d = 0; d < 4; ++d) o[d] = vzero;
  float rmax[8], rsum[8];
#pragma unroll
  for (int j = 0; j < 8; ++j) { rmax[j] = -__builtin_inff(); rsum[j] = 0.f; }

  __bf16* Pw = &Ps[wid][0][0];

  for (int kb = 0; kb <= qb; ++kb) {
    const int kbase = kb * 128;
    // prefetch next K/V block into cache (global_prefetch_b8)
    if (kb + 1 <= qb) {
      __builtin_prefetch(K + (size_t)(kbase + 128 + lane * 4) * DHEAD, 0, 1);
      __builtin_prefetch(V + (size_t)(lane * 2) * SEQ + kbase + 128, 0, 1);
    }
    // ---- scores: 8 column tiles x 2 k-steps = 16 WMMAs
    v8f sc[8];
#pragma unroll
    for (int f = 0; f < 8; ++f) {
      v8f c = vzero;
#pragma unroll
      for (int s = 0; s < 2; ++s) {
        const __bf16* p = K + (size_t)(kbase + f * 16 + l16) * DHEAD + s * 32 + half * 16;
        c = wmma_bf16(qf[s], frag_b(p), c);
      }
      sc[f] = c;
    }
    // ---- causal mask on the diagonal block (uniform branch -> EXEC stays full)
    if (kb == qb) {
#pragma unroll
      for (int f = 0; f < 8; ++f) {
        const int key = kbase + f * 16 + l16;
#pragma unroll
        for (int j = 0; j < 8; ++j) {
          const int qrow = qbase + j + half * 8;
          if (key > qrow) sc[f][j] = -__builtin_inff();
        }
      }
    }
    // ---- online softmax
    float mnew[8];
#pragma unroll
    for (int j = 0; j < 8; ++j) {
      float mx = rmax[j];
#pragma unroll
      for (int f = 0; f < 8; ++f) mx = fmaxf(mx, sc[f][j]);
      mnew[j] = redmax16(mx);
    }
    float psum[8];
#pragma unroll
    for (int j = 0; j < 8; ++j) psum[j] = 0.f;
#pragma unroll
    for (int f = 0; f < 8; ++f) {
#pragma unroll
      for (int j = 0; j < 8; ++j) {
        const float p = __expf(sc[f][j] - mnew[j]);
        psum[j] += p;
        Pw[(j + half * 8) * 136 + f * 16 + l16] = (__bf16)p;   // C-layout -> LDS
      }
    }
#pragma unroll
    for (int j = 0; j < 8; ++j) {
      const float s = redsum16(psum[j]);
      const float scale = __expf(rmax[j] - mnew[j]);           // first iter: exp(-inf)=0
      rsum[j] = rsum[j] * scale + s;
      rmax[j] = mnew[j];
#pragma unroll
      for (int d = 0; d < 4; ++d) o[d][j] *= scale;
    }
    // ---- P @ V : 4 k-steps x 4 dh tiles = 16 WMMAs (P re-read as A-frags)
#pragma unroll
    for (int s = 0; s < 4; ++s) {
      v16bf af = frag_a(&Pw[l16 * 136 + s * 32 + half * 16]);
#pragma unroll
      for (int d = 0; d < 4; ++d) {
        const __bf16* p = V + (size_t)(d * 16 + l16) * SEQ + kbase + s * 32 + half * 16;
        o[d] = wmma_bf16(af, frag_b(p), o[d]);
      }
    }
  }

  // ---- normalize + store y as bf16 in [b][s][h*64+dh] (GEMM-ready rows)
  const int b = bh >> 4, h = bh & 15;
#pragma unroll
  for (int d = 0; d < 4; ++d) {
#pragma unroll
    for (int j = 0; j < 8; ++j) {
      const int qrow = qbase + j + half * 8;
      const float val = o[d][j] / rsum[j];
      Yb[((size_t)(b * SEQ + qrow)) * D_MODEL + h * DHEAD + d * 16 + l16] = (__bf16)val;
    }
  }
}

// ---------------------------------------------------------------- launch
extern "C" void kernel_launch(void* const* d_in, const int* in_sizes, int n_in,
                              void* d_out, int out_size, void* d_ws, size_t ws_size,
                              hipStream_t stream) {
  (void)in_sizes; (void)n_in; (void)out_size; (void)ws_size;
  const float* x  = (const float*)d_in[0];
  const float* Wq = (const float*)d_in[1]; const float* bq = (const float*)d_in[2];
  const float* Wk = (const float*)d_in[3]; const float* bk = (const float*)d_in[4];
  const float* Wv = (const float*)d_in[5]; const float* bv = (const float*)d_in[6];
  const float* Wo = (const float*)d_in[7]; const float* bo = (const float*)d_in[8];

  __bf16* ws = (__bf16*)d_ws;
  const size_t XN = (size_t)ROWS * D_MODEL;      // 4,194,304
  const size_t WN = (size_t)D_MODEL * D_MODEL;   // 1,048,576
  __bf16* Xb  = ws;
  __bf16* WqT = Xb  + XN;
  __bf16* WkT = WqT + WN;
  __bf16* WvT = WkT + WN;
  __bf16* WoT = WvT + WN;
  __bf16* Qb  = WoT + WN;
  __bf16* Kb  = Qb  + XN;
  __bf16* Vt  = Kb  + XN;
  __bf16* Yb  = Vt  + XN;            // total ~48 MB of workspace

  cvt_bf16<<<(int)(XN / 4 / 256), 256, 0, stream>>>(x, Xb, (int)XN);
  cvt_bf16_T<<<(int)(WN / 256), 256, 0, stream>>>(Wq, WqT);
  cvt_bf16_T<<<(int)(WN / 256), 256, 0, stream>>>(Wk, WkT);
  cvt_bf16_T<<<(int)(WN / 256), 256, 0, stream>>>(Wv, WvT);
  cvt_bf16_T<<<(int)(WN / 256), 256, 0, stream>>>(Wo, WoT);

  dim3 gg(ROWS / 128, D_MODEL / 128);            // (32, 8)
  gemm_bf16<<<gg, 256, 0, stream>>>(Xb, WqT, bq, Qb, 0);
  gemm_bf16<<<gg, 256, 0, stream>>>(Xb, WkT, bk, Kb, 1);
  gemm_bf16<<<gg, 256, 0, stream>>>(Xb, WvT, bv, Vt, 2);

  attn<<<dim3(SEQ / 128, NBATCH * NH), 256, 0, stream>>>(Qb, Kb, Vt, Yb);

  gemm_bf16<<<gg, 256, 0, stream>>>(Yb, WoT, bo, d_out, 3);
}
